// QuantizedLinear_52879637348395
// MI455X (gfx1250) — matrix operand order repaired and checked by reading, not compiled, not run on hardware
//
#include <hip/hip_runtime.h>
#include <hip/hip_bf16.h>
#include <stdint.h>

// QuantizedLinear as a bf16-WMMA GEMM for gfx1250 (MI455X).
// out[M,N] = (x[M,K] @ wq[N,K]^T) * 0.01 + bq[N]*0.01,  M=N=K=4096.
// Compute-bound (AI ~720 FLOP/B) -> v_wmma_f32_16x16x32_bf16 core.
//
// Pre-pass (once, ~5us HBM): x f32->bf16, w i32->bf16 transposed to [K][N] in
// d_ws. GEMM staging is then pure copies on the CDNA5 async path:
// global_load_async_to_lds_b128 (ASYNCcnt) + s_wait_asynccnt, double-buffered,
// one barrier per k-tile, fully incremental addressing.
// Block tile 128m x 256n, 8 waves of 64x64 -> 16 WMMA per 8 ds_load_b128 (A) + 64 16-bit B loads,
// sized against ~8 cyc/WMMA implied by the CDNA5 matrix-rate quotes.

typedef __bf16 v16bf __attribute__((ext_vector_type(16)));
typedef __bf16 v8bf  __attribute__((ext_vector_type(8)));
typedef __bf16 v4bf  __attribute__((ext_vector_type(4)));
typedef float  v8f   __attribute__((ext_vector_type(8)));

#define TILE_M 128
#define TILE_N 256
#define TILE_K 32
#define XS_PITCH 40     // 32 + 8 pad   -> 80 B rows
#define WS_PITCH 272    // 256 + 16 pad -> 544 B rows
#define XBUF_STRIDE (TILE_M * XS_PITCH * 2)   // 10240 B
#define WBUF_STRIDE (TILE_K * WS_PITCH * 2)   // 17408 B
#define SCALE 0.01f

// ---- CDNA5 async copy primitives (inline asm: portable across toolchains) ----
__device__ __forceinline__ unsigned lds_off(const void* p) {
    return (unsigned)(uintptr_t)p;   // generic LDS pointer: offset in addr[31:0]
}
__device__ __forceinline__ void async_copy_b128(unsigned lds_byte_addr, unsigned long long gaddr) {
    asm volatile("global_load_async_to_lds_b128 %0, %1, off"
                 :: "v"(lds_byte_addr), "v"(gaddr)
                 : "memory");
}
__device__ __forceinline__ void wait_async_0() {
    asm volatile("s_wait_asynccnt 0x0" ::: "memory");
}

// =============================== pre-pass =====================================
__global__ __launch_bounds__(256)
void cvt_x_bf16(const float* __restrict__ x, __bf16* __restrict__ xb, long long n) {
    long long i = ((long long)blockIdx.x * 256 + threadIdx.x) * 8;
    if (i >= n) return;                       // n is a multiple of 8
    float4 a = *(const float4*)(x + i);
    float4 b = *(const float4*)(x + i + 4);
    v8bf t = { (__bf16)a.x, (__bf16)a.y, (__bf16)a.z, (__bf16)a.w,
               (__bf16)b.x, (__bf16)b.y, (__bf16)b.z, (__bf16)b.w };
    *(v8bf*)(xb + i) = t;
}

// wq[N][K] i32  ->  wt[K][N] bf16  (32x32 LDS tile transpose)
__global__ __launch_bounds__(256)
void cvt_w_t_bf16(const int* __restrict__ wq, __bf16* __restrict__ wt, int N, int K) {
    __shared__ __bf16 t[32][36];
    const int k0 = blockIdx.x * 32;
    const int n0 = blockIdx.y * 32;
    const int r  = threadIdx.x >> 3;
    const int c  = (threadIdx.x & 7) * 4;
    const int4 v = *(const int4*)(wq + (size_t)(n0 + r) * K + k0 + c);
    t[r][c + 0] = (__bf16)(float)v.x;
    t[r][c + 1] = (__bf16)(float)v.y;
    t[r][c + 2] = (__bf16)(float)v.z;
    t[r][c + 3] = (__bf16)(float)v.w;
    __syncthreads();
    v4bf o = { t[c + 0][r], t[c + 1][r], t[c + 2][r], t[c + 3][r] };
    *(v4bf*)(wt + (size_t)(k0 + r) * N + n0 + c) = o;
}

// ========================= main GEMM (async staging) ==========================
__global__ __launch_bounds__(256)
void qlinear_wmma_bf16_async(const __bf16* __restrict__ xb,   // [M][K]
                             const __bf16* __restrict__ wt,   // [K][N]
                             const int* __restrict__ bq,
                             float* __restrict__ out,
                             int M, int N, int K) {
    __shared__ __align__(16) __bf16 xs[2][TILE_M][XS_PITCH];  // A tile [m][k]
    __shared__ __align__(16) __bf16 ws[2][TILE_K][WS_PITCH];  // B tile [k][n]

    const int tid  = threadIdx.x;
    const int lane = tid & 31;
    const int wv   = tid >> 5;
    const int wm   = (wv & 1) * 64;       // 2 waves in m, 64 rows each
    const int wn   = (wv >> 1) * 64;      // 4 waves in n, 64 cols each
    const int m0   = blockIdx.y * TILE_M;
    const int n0   = blockIdx.x * TILE_N;

    const int halfSel = lane >> 4;
    const int am      = lane & 15;
    const int klo     = halfSel * 8;

    // ---- incremental async-copy state (2 x-chunks + 4 w-chunks per thread) ----
    // x tile: 128 rows x 4 chunks of 16B; w tile: 32 rows x 32 chunks of 16B.
    const int xi0 = tid * 2;
    unsigned long long gx[2];
    unsigned lx[2];
#pragma unroll
    for (int q = 0; q < 2; ++q) {
        const int i = xi0 + q, row = i >> 2, seg = i & 3;
        gx[q] = (unsigned long long)(uintptr_t)(xb + (size_t)(m0 + row) * K + seg * 8);
        lx[q] = lds_off(&xs[0][row][seg * 8]);
    }
    const int wi0 = tid * 4;
    unsigned long long gw[4];
    unsigned lw[4];
#pragma unroll
    for (int q = 0; q < 4; ++q) {
        const int i = wi0 + q, row = i >> 5, seg = i & 31;
        gw[q] = (unsigned long long)(uintptr_t)(wt + (size_t)row * N + n0 + seg * 8);
        lw[q] = lds_off(&ws[0][row][seg * 8]);
    }
    const unsigned long long GX_STEP = (unsigned long long)TILE_K * 2;       // 64 B
    const unsigned long long GW_STEP = (unsigned long long)TILE_K * N * 2;   // 256 KB

    auto issue_tile = [&](int cur) {     // issue 6 copies, then advance pointers
        const unsigned xadd = cur ? XBUF_STRIDE : 0;
        const unsigned wadd = cur ? WBUF_STRIDE : 0;
#pragma unroll
        for (int q = 0; q < 2; ++q) { async_copy_b128(lx[q] + xadd, gx[q]); gx[q] += GX_STEP; }
#pragma unroll
        for (int q = 0; q < 4; ++q) { async_copy_b128(lw[q] + wadd, gw[q]); gw[q] += GW_STEP; }
    };

    v8f acc[4][4];
#pragma unroll
    for (int mi = 0; mi < 4; ++mi)
#pragma unroll
        for (int ni = 0; ni < 4; ++ni)
            acc[mi][ni] = (v8f){0.f, 0.f, 0.f, 0.f, 0.f, 0.f, 0.f, 0.f};

    const int nk = K / TILE_K;
    issue_tile(0);

    for (int t = 0; t < nk; ++t) {
        const int cur = t & 1;
        wait_async_0();                        // this wave's chunks landed
        __syncthreads();                       // everyone's chunks landed
        if (t + 1 < nk)
            issue_tile(cur ^ 1);               // overlap next tile with WMMAs

        // B fragments: n = lane&15, elements = k (klo+0..7, 16+klo+0..7)
        v16bf bfrag[4];
#pragma unroll
        for (int ni = 0; ni < 4; ++ni) {
            const __bf16* bp = &ws[cur][klo][wn + ni * 16 + am];
#pragma unroll
            for (int s = 0; s < 8; ++s) {
                bfrag[ni][s]     = bp[s * WS_PITCH];
                bfrag[ni][8 + s] = bp[(16 + s) * WS_PITCH];
            }
        }

        // A fragments (ISA 16-bit A 16x32 layout) + 16 WMMAs
#pragma unroll
        for (int mi = 0; mi < 4; ++mi) {
            const __bf16* ap = &xs[cur][wm + mi * 16 + am][klo];
            v8bf a0 = *(const v8bf*)ap;
            v8bf a1 = *(const v8bf*)(ap + 16);
            v16bf a = __builtin_shufflevector(a0, a1,
                0, 1, 2, 3, 4, 5, 6, 7, 8, 9, 10, 11, 12, 13, 14, 15);
#pragma unroll
            for (int ni = 0; ni < 4; ++ni) {
                acc[mi][ni] = __builtin_amdgcn_wmma_f32_16x16x32_bf16(
                    false, a, false, bfrag[ni], (short)0, acc[mi][ni], false, false);
            }
        }
    }

    // epilogue: scale, bias, store (C/D: VGPR r -> M=r+8*halfSel, lane&15 -> N)
#pragma unroll
    for (int ni = 0; ni < 4; ++ni) {
        const int n  = n0 + wn + ni * 16 + am;
        const float bv = SCALE * (float)bq[n];
#pragma unroll
        for (int mi = 0; mi < 4; ++mi) {
            const int mbase = m0 + wm + mi * 16 + halfSel * 8;
            v8f c = acc[mi][ni];
#pragma unroll
            for (int r = 0; r < 8; ++r)
                out[(size_t)(mbase + r) * N + n] = c[r] * SCALE + bv;
        }
    }
}

// ==================== fallback: fused conversion GEMM =========================
#define FTILE_N 128
#define FWS_PITCH 136
__global__ __launch_bounds__(256)
void qlinear_wmma_bf16_fused(const float* __restrict__ x,
                             const int* __restrict__ wq,
                             const int* __restrict__ bq,
                             float* __restrict__ out,
                             int M, int N, int K) {
    __shared__ __align__(16) __bf16 xs[2][TILE_M][XS_PITCH];
    __shared__ __align__(16) __bf16 ws[2][TILE_K][FWS_PITCH];

    const int tid  = threadIdx.x;
    const int lane = tid & 31;
    const int wv   = tid >> 5;
    const int wm   = (wv & 1) * 64;
    const int wn   = (wv >> 1) * 32;
    const int m0   = blockIdx.y * TILE_M;
    const int n0   = blockIdx.x * FTILE_N;
    const int halfSel = lane >> 4;
    const int am      = lane & 15;
    const int klo     = halfSel * 8;
    const int xr = tid >> 3, xc = (tid & 7) * 4;
    const int wk = lane, wng = wv;

    float4 xv[4];
    int    wvr[16];
    auto load_tile = [&](int k0) {
#pragma unroll
        for (int p = 0; p < 4; ++p)
            xv[p] = *(const float4*)(x + (size_t)(m0 + p * 32 + xr) * K + k0 + xc);
        const int* wp = wq + (size_t)(n0 + wng * 16) * K + k0 + wk;
#pragma unroll
        for (int j = 0; j < 16; ++j)
            wvr[j] = wp[(size_t)j * K];
    };
    auto store_tile = [&](int cur) {
#pragma unroll
        for (int p = 0; p < 4; ++p) {
            v4bf t = { (__bf16)xv[p].x, (__bf16)xv[p].y, (__bf16)xv[p].z, (__bf16)xv[p].w };
            *(v4bf*)&xs[cur][p * 32 + xr][xc] = t;
        }
        v8bf w0, w1;
#pragma unroll
        for (int j = 0; j < 8; ++j) {
            w0[j] = (__bf16)(float)wvr[j];
            w1[j] = (__bf16)(float)wvr[8 + j];
        }
        *(v8bf*)&ws[cur][wk][wng * 16]     = w0;
        *(v8bf*)&ws[cur][wk][wng * 16 + 8] = w1;
    };

    v8f acc[4][2];
#pragma unroll
    for (int mi = 0; mi < 4; ++mi)
#pragma unroll
        for (int ni = 0; ni < 2; ++ni)
            acc[mi][ni] = (v8f){0.f, 0.f, 0.f, 0.f, 0.f, 0.f, 0.f, 0.f};

    const int nk = K / TILE_K;
    load_tile(0);
    for (int t = 0; t < nk; ++t) {
        const int cur = t & 1;
        store_tile(cur);
        __syncthreads();
        if (t + 1 < nk) load_tile((t + 1) * TILE_K);
        v16bf bfrag[2];
#pragma unroll
        for (int ni = 0; ni < 2; ++ni) {
            const __bf16* bp = &ws[cur][klo][wn + ni * 16 + am];
#pragma unroll
            for (int s = 0; s < 8; ++s) {
                bfrag[ni][s]     = bp[s * FWS_PITCH];
                bfrag[ni][8 + s] = bp[(16 + s) * FWS_PITCH];
            }
        }
#pragma unroll
        for (int mi = 0; mi < 4; ++mi) {
            const __bf16* ap = &xs[cur][wm + mi * 16 + am][klo];
            v8bf a0 = *(const v8bf*)ap;
            v8bf a1 = *(const v8bf*)(ap + 16);
            v16bf a = __builtin_shufflevector(a0, a1,
                0, 1, 2, 3, 4, 5, 6, 7, 8, 9, 10, 11, 12, 13, 14, 15);
#pragma unroll
            for (int ni = 0; ni < 2; ++ni)
                acc[mi][ni] = __builtin_amdgcn_wmma_f32_16x16x32_bf16(
                    false, a, false, bfrag[ni], (short)0, acc[mi][ni], false, false);
        }
    }
#pragma unroll
    for (int ni = 0; ni < 2; ++ni) {
        const int n  = n0 + wn + ni * 16 + am;
        const float bv = SCALE * (float)bq[n];
#pragma unroll
        for (int mi = 0; mi < 4; ++mi) {
            const int mbase = m0 + wm + mi * 16 + halfSel * 8;
            v8f c = acc[mi][ni];
#pragma unroll
            for (int r = 0; r < 8; ++r)
                out[(size_t)(mbase + r) * N + n] = c[r] * SCALE + bv;
        }
    }
}

// ================================ launcher ====================================
extern "C" void kernel_launch(void* const* d_in, const int* in_sizes, int n_in,
                              void* d_out, int out_size, void* d_ws, size_t ws_size,
                              hipStream_t stream) {
    (void)n_in; (void)out_size;
    const float* x   = (const float*)d_in[0];
    const int*   wq  = (const int*)d_in[1];   // harness widens int8 -> int32
    const int*   bq  = (const int*)d_in[2];
    float*       out = (float*)d_out;

    const int N = in_sizes[2];                // 4096
    const int K = in_sizes[1] / N;            // 4096
    const int M = in_sizes[0] / K;            // 4096 (= B*S)

    const size_t need = ((size_t)M * K + (size_t)K * N) * sizeof(__bf16);

    if (d_ws && ws_size >= need) {
        __bf16* xbuf = (__bf16*)d_ws;
        __bf16* wbuf = xbuf + (size_t)M * K;
        const long long nx = (long long)M * K;
        cvt_x_bf16<<<(unsigned)((nx / 8 + 255) / 256), 256, 0, stream>>>(x, xbuf, nx);
        dim3 gw(K / 32, N / 32);
        cvt_w_t_bf16<<<gw, 256, 0, stream>>>(wq, wbuf, N, K);
        dim3 grid(N / TILE_N, M / TILE_M);
        qlinear_wmma_bf16_async<<<grid, 256, 0, stream>>>(xbuf, wbuf, bq, out, M, N, K);
    } else {
        dim3 grid(N / FTILE_N, M / TILE_M);
        qlinear_wmma_bf16_fused<<<grid, 256, 0, stream>>>(x, wq, bq, out, M, N, K);
    }
}
